// MultiHeadSelfAttention_58978490909336
// MI455X (gfx1250) — matrix operand order repaired and checked by reading, not compiled, not run on hardware
//
#include <hip/hip_runtime.h>
#include <hip/hip_bf16.h>
#include <math.h>

#define D_MODEL 2048
#define NUM_HEADS 16
#define D_HEAD 64
#define SEQ 2048
#define BATCH 2
#define ROWS (BATCH * SEQ)   // 4096 token rows

typedef __attribute__((ext_vector_type(16))) __bf16 v16bf;
typedef __attribute__((ext_vector_type(8)))  float  v8f;

union Frag16 {
  v16bf v;
  uint4 u[2];
  __bf16 e[16];
};

// A-operand fragment (16x32, M x K): lanes 0-15 -> M=lane, K {0..7,16..23};
// lanes 16-31 -> M=lane-16, K {8..15,24..31}. Source row-major along K.
__device__ inline v16bf load_frag_a_g(const __bf16* __restrict__ base, int ld,
                                      int row0, int k0, int lane) {
  Frag16 f;
  const __bf16* p = base + (size_t)(row0 + (lane & 15)) * ld + k0 + ((lane >> 4) * 8);
  f.u[0] = *(const uint4*)(p);
  f.u[1] = *(const uint4*)(p + 16);
  return f.v;
}

// B-operand fragment (32x16, K x N): lanes 0-15 -> N=lane, K {0..7,16..23};
// lanes 16-31 -> N=lane-16, K {8..15,24..31}. Source holds column N as a contiguous
// row along K (i.e. we feed W row-major as B^T).
__device__ inline v16bf load_frag_b_g(const __bf16* __restrict__ base, int ld,
                                      int row0, int k0, int lane) {
  Frag16 f;
  const __bf16* p = base + (size_t)(row0 + (lane & 15)) * ld + k0 + ((lane >> 4) * 8);
  f.u[0] = *(const uint4*)(p);
  f.u[1] = *(const uint4*)(p + 16);
  return f.v;
}

// ---------------------------------------------------------------------------
// fp32 -> bf16 conversion, 4 elements per thread (b128 load, b64 store)
// ---------------------------------------------------------------------------
__global__ void cvt_f32_bf16_x4(const float4* __restrict__ s, ushort4* __restrict__ d, int n4) {
  int i = blockIdx.x * blockDim.x + threadIdx.x;
  if (i >= n4) return;
  float4 f = s[i];
  union { __bf16 e[4]; ushort4 u; } pk;
  pk.e[0] = (__bf16)f.x; pk.e[1] = (__bf16)f.y;
  pk.e[2] = (__bf16)f.z; pk.e[3] = (__bf16)f.w;
  d[i] = pk.u;
}

// ---------------------------------------------------------------------------
// GEMM: C[M,N] = A[M,K] (row-major) * B[N,K]^T (B row-major along K)
// Block: 256 threads = 8 waves; block tile 128(M) x 256(N); wave tile 64x64.
// 16 WMMAs per k-step vs 16 B128 loads (wmma:load = 1.0).
// ---------------------------------------------------------------------------
template <bool OUT_BF16>
__global__ __launch_bounds__(256) void gemm_xwt(const __bf16* __restrict__ A,
                                                const __bf16* __restrict__ B,
                                                void* __restrict__ Cout,
                                                int M, int N, int K) {
  const int lane = threadIdx.x & 31;
  const int wave = threadIdx.x >> 5;
  const int m0 = blockIdx.y * 128 + (wave & 1) * 64;
  const int n0 = blockIdx.x * 256 + (wave >> 1) * 64;

  v8f acc[4][4];
  for (int i = 0; i < 4; ++i)
    for (int j = 0; j < 4; ++j)
      for (int r = 0; r < 8; ++r) acc[i][j][r] = 0.0f;

#pragma unroll 1
  for (int k0 = 0; k0 < K; k0 += 32) {
    if (k0 + 64 < K) {
      __builtin_prefetch(A + (size_t)m0 * K + k0 + 64, 0, 1);
      __builtin_prefetch(B + (size_t)n0 * K + k0 + 64, 0, 1);
    }
    v16bf a[4], b[4];
#pragma unroll
    for (int t = 0; t < 4; ++t) {
      a[t] = load_frag_a_g(A, K, m0 + t * 16, k0, lane);
      b[t] = load_frag_b_g(B, K, n0 + t * 16, k0, lane);
    }
#pragma unroll
    for (int mt = 0; mt < 4; ++mt)
#pragma unroll
      for (int nt = 0; nt < 4; ++nt)
        acc[mt][nt] = __builtin_amdgcn_wmma_f32_16x16x32_bf16(
            false, a[mt], false, b[nt], (short)0, acc[mt][nt], false, false);
  }

  const int lh = lane >> 4, lc = lane & 15;
  for (int mt = 0; mt < 4; ++mt)
    for (int nt = 0; nt < 4; ++nt)
      for (int r = 0; r < 8; ++r) {
        int row = m0 + mt * 16 + r + lh * 8;
        int col = n0 + nt * 16 + lc;
        float v = acc[mt][nt][r];
        if (OUT_BF16) ((__bf16*)Cout)[(size_t)row * N + col] = (__bf16)v;
        else          ((float*)Cout)[(size_t)row * N + col] = v;
      }
}

// ---------------------------------------------------------------------------
// RoPE in place on bf16 Q and K (fp32 math). One thread per (token, head, pair).
// ---------------------------------------------------------------------------
__global__ void rope_qk(__bf16* __restrict__ q, __bf16* __restrict__ k, int npairs) {
  int idx = blockIdx.x * blockDim.x + threadIdx.x;
  if (idx >= npairs) return;
  int row = idx >> 10;          // token row (b*S + s), 1024 pairs per row
  int p   = idx & 1023;
  int h   = p >> 5;
  int i   = p & 31;
  int s   = row & (SEQ - 1);
  // inv_freq = 10000^(-2i/64) = exp(-i * ln(10000)/32)
  float inv = __expf(-(float)i * 0.28782313662425572f);
  float ang = (float)s * inv;
  float sn, cs;
  __sincosf(ang, &sn, &cs);
  size_t base = (size_t)row * D_MODEL + h * D_HEAD + 2 * i;
  float qe = (float)q[base], qo = (float)q[base + 1];
  q[base]     = (__bf16)(qe * cs - qo * sn);
  q[base + 1] = (__bf16)(qe * sn + qo * cs);
  float ke = (float)k[base], ko = (float)k[base + 1];
  k[base]     = (__bf16)(ke * cs - ko * sn);
  k[base + 1] = (__bf16)(ke * sn + ko * cs);
}

// ---------------------------------------------------------------------------
// Causal flash attention. One wave (one 32-thread block) per (b, h, 16-query
// tile). 32 kv positions per iteration; online softmax in fp32; Q*K^T and P*V
// via bf16 WMMA. P and V are staged through LDS to build WMMA fragments.
// ---------------------------------------------------------------------------
__global__ __launch_bounds__(32) void attn_fwd(const __bf16* __restrict__ Q,
                                               const __bf16* __restrict__ Kb,
                                               const __bf16* __restrict__ V,
                                               __bf16* __restrict__ O) {
  __shared__ __bf16 Pl[16 * 32];   // P tile, row-major, ld=32
  __shared__ __bf16 Vl[32 * 64];   // V tile, row-major [kv][d], ld=64
  const int lane = threadIdx.x;
  const int lh = lane >> 4, lc = lane & 15;
  const int bid = blockIdx.x;
  const int qt = bid & 127;
  const int h  = (bid >> 7) & 15;
  const int b  = bid >> 11;
  const int q0 = qt * 16;

  const __bf16* Qh = Q  + (size_t)b * SEQ * D_MODEL + h * D_HEAD;
  const __bf16* Kh = Kb + (size_t)b * SEQ * D_MODEL + h * D_HEAD;
  const __bf16* Vh = V  + (size_t)b * SEQ * D_MODEL + h * D_HEAD;

  v16bf qa0 = load_frag_a_g(Qh, D_MODEL, q0, 0,  lane);
  v16bf qa1 = load_frag_a_g(Qh, D_MODEL, q0, 32, lane);

  float mrow[8], lrow[8];
  v8f acc[4];
  for (int r = 0; r < 8; ++r) { mrow[r] = -3.0e38f; lrow[r] = 0.0f; }
  for (int dc = 0; dc < 4; ++dc)
    for (int r = 0; r < 8; ++r) acc[dc][r] = 0.0f;

  for (int kb = 0; kb <= q0 + 15; kb += 32) {
    // ---- scores: two 16x16 tiles over this 32-kv window -------------------
    float st[2][8];
#pragma unroll
    for (int t = 0; t < 2; ++t) {
      v16bf kf0 = load_frag_b_g(Kh, D_MODEL, kb + t * 16, 0,  lane);
      v16bf kf1 = load_frag_b_g(Kh, D_MODEL, kb + t * 16, 32, lane);
      v8f s;
      for (int r = 0; r < 8; ++r) s[r] = 0.0f;
      s = __builtin_amdgcn_wmma_f32_16x16x32_bf16(false, qa0, false, kf0, (short)0, s, false, false);
      s = __builtin_amdgcn_wmma_f32_16x16x32_bf16(false, qa1, false, kf1, (short)0, s, false, false);
#pragma unroll
      for (int r = 0; r < 8; ++r) {
        int qi = q0 + r + lh * 8;
        int ki = kb + t * 16 + lc;
        st[t][r] = (ki <= qi) ? s[r] * 0.125f : -3.0e38f;
      }
    }

    // ---- stage V tile (32 kv rows x 64 dims) into LDS, row-major ----------
    {
      const __bf16* vr = Vh + (size_t)(kb + lane) * D_MODEL;
#pragma unroll
      for (int j = 0; j < 8; ++j)
        *(uint4*)&Vl[lane * 64 + j * 8] = *(const uint4*)(vr + j * 8);
    }

    // ---- online softmax (rows live across 16-lane half-groups) ------------
    float pnew[2][8];
#pragma unroll
    for (int r = 0; r < 8; ++r) {
      float tm = fmaxf(st[0][r], st[1][r]);
      for (int o = 8; o > 0; o >>= 1) tm = fmaxf(tm, __shfl_xor(tm, o, 32));
      float mn = fmaxf(mrow[r], tm);
      float sc = __expf(mrow[r] - mn);
      mrow[r] = mn;
      lrow[r] *= sc;
      for (int dc = 0; dc < 4; ++dc) acc[dc][r] *= sc;
      float p0 = __expf(st[0][r] - mn);
      float p1 = __expf(st[1][r] - mn);
      pnew[0][r] = p0; pnew[1][r] = p1;
      float ps = p0 + p1;
      for (int o = 8; o > 0; o >>= 1) ps += __shfl_xor(ps, o, 32);
      lrow[r] += ps;
    }

    // ---- store P (C-layout -> row-major LDS) ------------------------------
#pragma unroll
    for (int t = 0; t < 2; ++t)
#pragma unroll
      for (int r = 0; r < 8; ++r)
        Pl[(r + lh * 8) * 32 + t * 16 + lc] = (__bf16)pnew[t][r];
    __syncthreads();   // single-wave workgroup: cheap DScnt fence

    // ---- P A-fragment from LDS -------------------------------------------
    Frag16 fp;
    {
      const __bf16* p = &Pl[(lane & 15) * 32 + (lane >> 4) * 8];
      fp.u[0] = *(const uint4*)(p);
      fp.u[1] = *(const uint4*)(p + 16);
    }

    // ---- P(16x32) * V(32x64) via 4 WMMAs ---------------------------------
#pragma unroll
    for (int dc = 0; dc < 4; ++dc) {
      Frag16 fb;
      int d = dc * 16 + lc;
      int kbase = lh * 8;
#pragma unroll
      for (int j = 0; j < 16; ++j) fb.e[j] = Vl[(kbase + j + (j >> 3) * 8) * 64 + d];
      acc[dc] = __builtin_amdgcn_wmma_f32_16x16x32_bf16(false, fp.v, false, fb.v, (short)0, acc[dc], false, false);
    }
    __syncthreads();
  }

  // ---- epilogue: divide by row sums, write bf16 ---------------------------
  for (int dc = 0; dc < 4; ++dc)
    for (int r = 0; r < 8; ++r) {
      int row = q0 + r + lh * 8;
      int col = dc * 16 + lc;
      float o = acc[dc][r] / lrow[r];
      O[(size_t)(b * SEQ + row) * D_MODEL + h * D_HEAD + col] = (__bf16)o;
    }
}

// ---------------------------------------------------------------------------
// Host launcher
// ---------------------------------------------------------------------------
extern "C" void kernel_launch(void* const* d_in, const int* in_sizes, int n_in,
                              void* d_out, int out_size, void* d_ws, size_t ws_size,
                              hipStream_t stream) {
  (void)in_sizes; (void)n_in; (void)out_size; (void)ws_size;
  const float* x   = (const float*)d_in[0];
  const float* w_q = (const float*)d_in[1];
  const float* w_k = (const float*)d_in[2];
  const float* w_v = (const float*)d_in[3];
  const float* w_o = (const float*)d_in[4];

  char* ws = (char*)d_ws;
  size_t off = 0;
  auto carve = [&](size_t bytes) -> char* {
    char* p = ws + off;
    off += (bytes + 255) & ~(size_t)255;
    return p;
  };
  const size_t nx = (size_t)ROWS * D_MODEL;       // 8.4M
  const size_t nw = (size_t)D_MODEL * D_MODEL;    // 4.2M
  __bf16* xb  = (__bf16*)carve(nx * 2);
  __bf16* wqb = (__bf16*)carve(nw * 2);
  __bf16* wkb = (__bf16*)carve(nw * 2);
  __bf16* wvb = (__bf16*)carve(nw * 2);
  __bf16* wob = (__bf16*)carve(nw * 2);
  __bf16* qb  = (__bf16*)carve(nx * 2);
  __bf16* kb  = (__bf16*)carve(nx * 2);
  __bf16* vb  = (__bf16*)carve(nx * 2);
  __bf16* ab  = (__bf16*)carve(nx * 2);   // attention output (token-major)

  // 1) fp32 -> bf16 conversions (vectorized x4)
  cvt_f32_bf16_x4<<<(int)((nx / 4 + 255) / 256), 256, 0, stream>>>((const float4*)x,   (ushort4*)xb,  (int)(nx / 4));
  cvt_f32_bf16_x4<<<(int)((nw / 4 + 255) / 256), 256, 0, stream>>>((const float4*)w_q, (ushort4*)wqb, (int)(nw / 4));
  cvt_f32_bf16_x4<<<(int)((nw / 4 + 255) / 256), 256, 0, stream>>>((const float4*)w_k, (ushort4*)wkb, (int)(nw / 4));
  cvt_f32_bf16_x4<<<(int)((nw / 4 + 255) / 256), 256, 0, stream>>>((const float4*)w_v, (ushort4*)wvb, (int)(nw / 4));
  cvt_f32_bf16_x4<<<(int)((nw / 4 + 255) / 256), 256, 0, stream>>>((const float4*)w_o, (ushort4*)wob, (int)(nw / 4));

  // 2) Q/K/V projections: (4096x2048) = X * W^T via bf16 WMMA
  dim3 gg(D_MODEL / 256, ROWS / 128), gb(256);
  gemm_xwt<true><<<gg, gb, 0, stream>>>(xb, wqb, qb, ROWS, D_MODEL, D_MODEL);
  gemm_xwt<true><<<gg, gb, 0, stream>>>(xb, wkb, kb, ROWS, D_MODEL, D_MODEL);
  gemm_xwt<true><<<gg, gb, 0, stream>>>(xb, wvb, vb, ROWS, D_MODEL, D_MODEL);

  // 3) RoPE on Q and K (in place)
  int npairs = ROWS * (D_MODEL / 2);               // one thread per even/odd pair
  rope_qk<<<(npairs + 255) / 256, 256, 0, stream>>>(qb, kb, npairs);

  // 4) causal flash attention, one wave per (b, h, 16-query tile)
  attn_fwd<<<BATCH * NUM_HEADS * (SEQ / 16), 32, 0, stream>>>(qb, kb, vb, ab);

  // 5) output projection to fp32 d_out
  gemm_xwt<false><<<gg, gb, 0, stream>>>(ab, wob, d_out, ROWS, D_MODEL, D_MODEL);
}
